// MultiHeadAttention_28295244546246
// MI455X (gfx1250) — matrix operand order repaired and checked by reading, not compiled, not run on hardware
//
#include <hip/hip_runtime.h>

// ---------------------------------------------------------------------------
// MultiHeadAttention (GAT-style) for MI455X / gfx1250.
//
// Math: logits[h,i,j] = (x@W1^T)[i,h] + (x@W2^T)[j,h] + b[h].
// Softmax over j is shift-invariant => attn[h,i,:] = softmax_j(s[h,:]),
// s[h,j] = dot(x[j], W2[h]).  W1 and b cancel exactly. Output rows are all
// identical: out[i,:] = leaky_relu( mean_h ( sum_j a[h,j] * x[j,:] ) ).
//
// Cost: read x twice (8 MB) + write out (4 MB) => HBM-bound, ~0.5us @ 23.3TB/s.
// The attn-apply GEMM (4x4096 @ 4096x256) runs on v_wmma_f32_16x16x32_f16.
// ---------------------------------------------------------------------------

typedef __attribute__((ext_vector_type(16))) _Float16 v16h;
typedef __attribute__((ext_vector_type(8)))  float    v8f;

#define NN 4096
#define FF 256
#define HH 4

// ws layout (floats):
//   s/e   : [4][4096]   offset 0      (scores, overwritten with exp values)
//   Z     : [4]         offset 16384  (softmax denominators)
//   u     : [256]       offset 16388  (final row vector, 16B aligned)
//   vpart : [32][4][256] offset 16644 (per-block WMMA partial sums)
#define WS_Z     16384
#define WS_U     16388
#define WS_VPART 16644

// ---- Kernel 1: s[h][j] = dot(x[j,:], W2[h,:]) --------------------------------
// 512 blocks x 256 threads = 8 waves/block, one j-row per wave.
// Lane reads 8 contiguous f32 (two float4) -> full 1KB row coalesced per wave.
__global__ void mha_scores(const float* __restrict__ x,
                           const float* __restrict__ W,
                           float* __restrict__ s) {
    const int lane = threadIdx.x & 31;
    const int wave = threadIdx.x >> 5;
    const int j    = blockIdx.x * 8 + wave;

    const float4* xr = (const float4*)(x + (size_t)j * FF + lane * 8);
    const float4 a0 = xr[0], a1 = xr[1];

#pragma unroll
    for (int h = 0; h < HH; ++h) {
        // W2[h][k] = W[h*512 + 256 + k]
        const float4* wr = (const float4*)(W + h * 512 + 256 + lane * 8);
        const float4 w0 = wr[0], w1 = wr[1];
        float p = a0.x*w0.x + a0.y*w0.y + a0.z*w0.z + a0.w*w0.w
                + a1.x*w1.x + a1.y*w1.y + a1.z*w1.z + a1.w*w1.w;
#pragma unroll
        for (int off = 16; off > 0; off >>= 1) p += __shfl_xor(p, off, 32);
        if (lane == 0) s[h * NN + j] = p;
    }
}

// ---- Kernel 2: per-head softmax over 4096 scores (one workgroup) ------------
// Deterministic tree reductions in LDS: max, then exp+sum. Overwrites s with e.
__global__ void mha_softmax(float* __restrict__ s, float* __restrict__ Z) {
    __shared__ float red[1024];
    const int t = threadIdx.x;
    const int h = t >> 8;        // 0..3
    const int c = t & 255;       // 0..255
    float* sh = s + h * NN;

    float mx = -3.402823466e+38f;
#pragma unroll
    for (int k = 0; k < 16; ++k) mx = fmaxf(mx, sh[c + (k << 8)]);
    red[t] = mx;
    __syncthreads();
#pragma unroll
    for (int off = 128; off > 0; off >>= 1) {
        if (c < off) red[t] = fmaxf(red[t], red[t + off]);
        __syncthreads();
    }
    const float mh = red[h << 8];

    float sum = 0.f;
#pragma unroll
    for (int k = 0; k < 16; ++k) {
        const int idx = c + (k << 8);
        const float e = __expf(sh[idx] - mh);
        sh[idx] = e;
        sum += e;
    }
    __syncthreads();              // all reads of red (mh) done before reuse
    red[t] = sum;
    __syncthreads();
#pragma unroll
    for (int off = 128; off > 0; off >>= 1) {
        if (c < off) red[t] += red[t + off];
        __syncthreads();
    }
    if (c == 0) Z[h] = red[t];
}

// ---- Kernel 3: WMMA attn-apply: vpart[b][h][f] = sum_{j in block b} e[h][j]*x[j][f]
// 32 blocks x 8 waves. Block b owns j in [b*128, b*128+128). Wave w owns
// f-tiles [w*32, w*32+16) and [w*32+16, w*32+32). D = A(16x32 f16) x B(32x16 f16)
// accumulated in f32 over 4 K-chunks. A rows 0..3 = heads, rows 4..15 = zero.
__global__ void mha_apply(const float* __restrict__ x,
                          const float* __restrict__ e,
                          float* __restrict__ vpart) {
    const int lane = threadIdx.x & 31;
    const int wave = threadIdx.x >> 5;
    const int half = lane >> 4;          // 0: lanes 0-15, 1: lanes 16-31
    const int lm   = lane & 15;
    const int jbase = blockIdx.x * 128;
    const int f0    = wave * 32;

    v8f acc0 = {0.f, 0.f, 0.f, 0.f, 0.f, 0.f, 0.f, 0.f};
    v8f acc1 = acc0;

#pragma unroll
    for (int kc = 0; kc < 4; ++kc) {
        const int j0 = jbase + kc * 32;

        // A fragment: lane M=lm; elem i -> K = half*16 + i
        v16h afrag;
#pragma unroll
        for (int i = 0; i < 16; ++i) afrag[i] = (_Float16)0.f;
        if (lm < HH) {
            const float* ep = e + lm * NN + j0 + half * 16;
#pragma unroll
            for (int i = 0; i < 8; ++i) afrag[i]     = (_Float16)ep[i];
#pragma unroll
            for (int i = 0; i < 8; ++i) afrag[8 + i] = (_Float16)ep[8 + i];
        }

        // B fragments: lane N=lm; elem i -> K = half*16 + i. Coalesced across
        // the 16-lane half (consecutive f at fixed j).
        const float* xp = x + (size_t)(j0 + half * 16) * FF + lm;
        v16h b0, b1;
#pragma unroll
        for (int i = 0; i < 16; ++i) b0[i] = (_Float16)xp[(size_t)i * FF + f0];
#pragma unroll
        for (int i = 0; i < 16; ++i) b1[i] = (_Float16)xp[(size_t)i * FF + f0 + 16];

        acc0 = __builtin_amdgcn_wmma_f32_16x16x32_f16(
                   false, afrag, false, b0, (short)0, acc0, false, false);
        acc1 = __builtin_amdgcn_wmma_f32_16x16x32_f16(
                   false, afrag, false, b1, (short)0, acc1, false, false);
    }

    // D layout: acc[r] at lane L is D[M = r + 8*half][N = lm]; heads are M<4.
    if (half == 0) {
        float* vp = vpart + (size_t)blockIdx.x * (HH * FF);
#pragma unroll
        for (int r = 0; r < HH; ++r) {
            vp[r * FF + f0 + lm]      = acc0[r];
            vp[r * FF + f0 + 16 + lm] = acc1[r];
        }
    }
}

// ---- Kernel 4: fixed-order reduce of partials + normalize + mean + leaky ----
__global__ void mha_reduce(const float* __restrict__ vpart,
                           const float* __restrict__ Z,
                           float* __restrict__ u) {
    const int f = threadIdx.x;
    float acc = 0.f;
#pragma unroll
    for (int h = 0; h < HH; ++h) {
        float ph = 0.f;
        for (int b = 0; b < 32; ++b) ph += vpart[(b * HH + h) * FF + f];
        acc += ph / Z[h];
    }
    acc *= (1.f / HH);
    u[f] = acc > 0.f ? acc : 0.2f * acc;   // leaky_relu, slope 0.2
}

// ---- Kernel 5: broadcast the 256-float row to all 4096 rows (float4 stores) -
__global__ void mha_bcast(const float4* __restrict__ u4, float4* __restrict__ out4) {
    const int idx = blockIdx.x * blockDim.x + threadIdx.x;  // 0..262143
    out4[idx] = u4[idx & 63];                               // 256 f32 = 64 float4
}

extern "C" void kernel_launch(void* const* d_in, const int* in_sizes, int n_in,
                              void* d_out, int out_size, void* d_ws, size_t ws_size,
                              hipStream_t stream) {
    (void)in_sizes; (void)n_in; (void)out_size; (void)ws_size;
    const float* x = (const float*)d_in[0];
    const float* W = (const float*)d_in[1];
    // d_in[2] (b) and W[:, :256] (W1) cancel in the softmax and are unused.
    float* out = (float*)d_out;
    float* ws  = (float*)d_ws;

    float* s     = ws;             // [4][4096] scores -> exp values
    float* Z     = ws + WS_Z;      // [4]
    float* u     = ws + WS_U;      // [256]
    float* vpart = ws + WS_VPART;  // [32][4][256]

    mha_scores <<<512,  256, 0, stream>>>(x, W, s);
    mha_softmax<<<1,   1024, 0, stream>>>(s, Z);
    mha_apply  <<<32,   256, 0, stream>>>(x, s, vpart);
    mha_reduce <<<1,    256, 0, stream>>>(vpart, Z, u);
    mha_bcast  <<<1024, 256, 0, stream>>>((const float4*)u, (float4*)out);
}
